// CfC_67542655697152
// MI455X (gfx1250) — compile-verified, matrix operand order checked by reading, not run on hardware
//
#include <hip/hip_runtime.h>
#include <hip/hip_bf16.h>
#include <math.h>

// ---------------- types ----------------
typedef __attribute__((ext_vector_type(16))) __bf16 v16bf;
typedef __attribute__((ext_vector_type(8)))  __bf16 v8bf;
typedef __attribute__((ext_vector_type(8)))  float  v8f;

// ---------------- problem dims ----------------
#define B_   64
#define T_   512
#define I_   256
#define H_   512
#define BU_  1024
#define KZ_  (I_ + H_)          // 768 = concat(x, h)

#define NWG_     32             // persistent grid: 32 workgroups
#define THREADS_ 256            // 8 waves each -> 256 waves total

// ---------------- workspace layout (bytes) ----------------
#define WS_BAR   0u
#define WS_W1    256u                               // [BU_][KZ_] bf16
#define WS_W2    (WS_W1  + 2u*BU_*KZ_)              // [BU_][BU_] bf16
#define WS_WF1   (WS_W2  + 2u*BU_*BU_)              // [H_][BU_] bf16
#define WS_WF2   (WS_WF1 + 2u*H_*BU_)
#define WS_WTA   (WS_WF2 + 2u*H_*BU_)
#define WS_WTB   (WS_WTA + 2u*H_*BU_)
#define WS_Z0    (WS_WTB + 2u*H_*BU_)               // [B_][KZ_] bf16 (concat buffer)
#define WS_Z1    (WS_Z0  + 2u*B_*KZ_)               // [B_][BU_] bf16
#define WS_Z2    (WS_Z1  + 2u*B_*BU_)               // [B_][BU_] bf16
// total ~8.2 MB (all L2-resident)

// ---------------- branchless fast activations (v_exp_f32 / v_rcp_f32) ----------------
#define LOG2E_ 1.4426950408889634f
__device__ __forceinline__ float fast_tanh(float x) {
  // tanh(x) = 1 - 2/(1 + exp2(2*log2e*x)); saturates correctly: x->-inf => -1, x->+inf => +1
  float e = __builtin_amdgcn_exp2f(x * (2.0f * LOG2E_));
  return 1.0f - 2.0f * __builtin_amdgcn_rcpf(e + 1.0f);
}
__device__ __forceinline__ float fast_sigmoid(float x) {
  return __builtin_amdgcn_rcpf(1.0f + __builtin_amdgcn_exp2f(-LOG2E_ * x));
}

// ---------------- device-wide barrier (sense via generation counter) ----------------
__device__ __forceinline__ void grid_sync(unsigned* cnt, volatile unsigned* gen) {
  __threadfence();               // release: flush this thread's stores to device scope
  __syncthreads();
  if (threadIdx.x == 0) {
    unsigned g = *gen;
    if (atomicAdd(cnt, 1u) == gridDim.x - 1) {
      atomicExch(cnt, 0u);
      __threadfence();
      atomicExch((unsigned*)gen, g + 1u);
    } else {
      while (*gen == g) { __builtin_amdgcn_s_sleep(1); }
    }
  }
  __syncthreads();
  __threadfence();               // acquire: drop stale WGP$ lines
}

// ---------------- bf16 16x(K=32) tile load matching WMMA A/B VGPR layout ----------------
// lanes 0-15: row r0+lane,    K = {0..7, 16..23} of the 32-slice
// lanes 16-31: row r0+lane-16, K = {8..15, 24..31}
// -> two aligned 16B loads per lane (global_load_b128)
__device__ __forceinline__ v16bf load_tile16(const __bf16* base, int ld, int r0, int k0) {
  const int lane = threadIdx.x & 31;
  const int half = lane >> 4;
  const int r    = r0 + (lane & 15);
  const __bf16* p = base + (size_t)r * ld + k0 + (half << 3);
  v8bf lo = *(const v8bf*)(p);
  v8bf hi = *(const v8bf*)(p + 16);
  v16bf v;
#pragma unroll
  for (int i = 0; i < 8; ++i) { v[i] = lo[i]; v[i + 8] = hi[i]; }
  return v;
}

// 16x16 output tile: A:[M,K] row-major, B:[N,K] row-major (i.e. X @ W^T)
__device__ __forceinline__ v8f gemm_tile(const __bf16* A, int lda,
                                         const __bf16* W, int ldw,
                                         int m0, int n0, int K) {
  v8f acc = {};
#pragma unroll 4
  for (int k0 = 0; k0 < K; k0 += 32) {
    v16bf a = load_tile16(A, lda, m0, k0);
    v16bf b = load_tile16(W, ldw, n0, k0);
    acc = __builtin_amdgcn_wmma_f32_16x16x32_bf16(false, a, false, b, (short)0, acc,
                                                  false, false);
  }
  return acc;
}

// C/D layout: VGPR v holds row m0 + v + 8*half, column n0 + (lane&15)
__device__ __forceinline__ void store_lecun_bf16(v8f acc, __bf16* Z, int ldz,
                                                 int m0, int n0, const float* bias) {
  const int lane = threadIdx.x & 31;
  const int half = lane >> 4;
  const int nc   = n0 + (lane & 15);
  const float bv = bias[nc];
#pragma unroll
  for (int v = 0; v < 8; ++v) {
    float val = 1.7159f * fast_tanh(0.666f * (acc[v] + bv));
    Z[(size_t)(m0 + v + 8 * half) * ldz + nc] = (__bf16)val;
  }
}

// ---------------- prep: weights f32->bf16, seed Z0 = [x[:,0,:], h0=0], reset barrier --------
__device__ __forceinline__ void conv_bf16(const float* s, __bf16* d, int n, int gid, int nth) {
  for (int i = gid; i < n; i += nth) d[i] = (__bf16)s[i];
}

__global__ void cfc_prep(const float* __restrict__ x,
                         const float* __restrict__ wb1, const float* __restrict__ wb2,
                         const float* __restrict__ wff1, const float* __restrict__ wff2,
                         const float* __restrict__ wta, const float* __restrict__ wtb,
                         unsigned char* __restrict__ ws) {
  const int gid = blockIdx.x * blockDim.x + threadIdx.x;
  const int nth = gridDim.x * blockDim.x;
  if (gid == 0) {
    unsigned* bar = (unsigned*)(ws + WS_BAR);
    bar[0] = 0u;  // arrival counter
    bar[1] = 0u;  // generation
  }
  conv_bf16(wb1,  (__bf16*)(ws + WS_W1),  BU_ * KZ_, gid, nth);
  conv_bf16(wb2,  (__bf16*)(ws + WS_W2),  BU_ * BU_, gid, nth);
  conv_bf16(wff1, (__bf16*)(ws + WS_WF1), H_ * BU_,  gid, nth);
  conv_bf16(wff2, (__bf16*)(ws + WS_WF2), H_ * BU_,  gid, nth);
  conv_bf16(wta,  (__bf16*)(ws + WS_WTA), H_ * BU_,  gid, nth);
  conv_bf16(wtb,  (__bf16*)(ws + WS_WTB), H_ * BU_,  gid, nth);
  __bf16* Z0 = (__bf16*)(ws + WS_Z0);
  for (int e = gid; e < B_ * I_; e += nth) {           // x part, t = 0
    int b = e >> 8, c = e & (I_ - 1);
    Z0[b * KZ_ + c] = (__bf16)x[((size_t)b * T_ + 0) * I_ + c];
  }
  for (int e = gid; e < B_ * H_; e += nth) {           // h0 = 0
    int b = e >> 9, c = e & (H_ - 1);
    Z0[b * KZ_ + I_ + c] = (__bf16)0.0f;
  }
}

// ---------------- persistent recurrence kernel: 3 grid syncs per timestep ----------------
__global__ __launch_bounds__(THREADS_) void cfc_persistent(
    const float* __restrict__ x,
    const float* __restrict__ bb1, const float* __restrict__ bb2,
    const float* __restrict__ bff1, const float* __restrict__ bff2,
    const float* __restrict__ bta, const float* __restrict__ btb,
    float* __restrict__ out, unsigned char* __restrict__ ws) {

  unsigned* bcnt = (unsigned*)(ws + WS_BAR);
  volatile unsigned* bgen = (volatile unsigned*)(ws + WS_BAR) + 1;

  const __bf16* W1  = (const __bf16*)(ws + WS_W1);
  const __bf16* W2  = (const __bf16*)(ws + WS_W2);
  const __bf16* WF1 = (const __bf16*)(ws + WS_WF1);
  const __bf16* WF2 = (const __bf16*)(ws + WS_WF2);
  const __bf16* WTA = (const __bf16*)(ws + WS_WTA);
  const __bf16* WTB = (const __bf16*)(ws + WS_WTB);

  __bf16* Z0 = (__bf16*)(ws + WS_Z0);
  __bf16* Z1 = (__bf16*)(ws + WS_Z1);
  __bf16* Z2 = (__bf16*)(ws + WS_Z2);

  // split-K reduction scratch: [pair][head][v][lane] f32, conflict-free lane-major
  __shared__ float red[4][4][8][32];

  const int lane = threadIdx.x & 31;
  const int wv   = threadIdx.x >> 5;                                  // wave in WG: 0..7
  const int wid  = blockIdx.x * (THREADS_ >> 5) + wv;                 // 0..255
  const int gid  = blockIdx.x * THREADS_ + threadIdx.x;               // 0..8191

  // phase C pairing: 4 pairs/WG, pair p owns one 16x16 tile of H; kh = K-half
  const int p    = wv >> 1;
  const int kh   = wv & 1;
  const int tile = blockIdx.x * 4 + p;                                // 0..127
  const int cm0  = (tile >> 5) << 4;                                  // batch-tile row
  const int cn0  = (tile & 31) << 4;                                  // H column tile

  for (int t = 0; t < T_; ++t) {
    // ---- phase A: Z1 = lecun(Z0 @ W1^T + bb1), 256 tiles, K=768 ----
    {
      const int m0 = (wid >> 6) << 4;
      const int n0 = (wid & 63) << 4;
      v8f acc = gemm_tile(Z0, KZ_, W1, KZ_, m0, n0, KZ_);
      store_lecun_bf16(acc, Z1, BU_, m0, n0, bb1);
    }
    grid_sync(bcnt, bgen);

    // ---- phase B: Z2 = lecun(Z1 @ W2^T + bb2), 256 tiles, K=1024 ----
    {
      const int m0 = (wid >> 6) << 4;
      const int n0 = (wid & 63) << 4;
      v8f acc = gemm_tile(Z1, BU_, W2, BU_, m0, n0, BU_);
      store_lecun_bf16(acc, Z2, BU_, m0, n0, bb2);
    }
    grid_sync(bcnt, bgen);

    // ---- phase C (fused heads + blend): wave pair splits K, 4 heads share each A-tile ----
    {
      v8f acc[4] = {{}, {}, {}, {}};
      const int kbase = kh * (BU_ / 2);
#pragma unroll 2
      for (int k0 = 0; k0 < BU_ / 2; k0 += 32) {
        v16bf a  = load_tile16(Z2, BU_, cm0, kbase + k0);   // 1 A-load feeds 4 WMMAs
        v16bf b0 = load_tile16(WF1, BU_, cn0, kbase + k0);
        acc[0] = __builtin_amdgcn_wmma_f32_16x16x32_bf16(false, a, false, b0, (short)0,
                                                         acc[0], false, false);
        v16bf b1 = load_tile16(WF2, BU_, cn0, kbase + k0);
        acc[1] = __builtin_amdgcn_wmma_f32_16x16x32_bf16(false, a, false, b1, (short)0,
                                                         acc[1], false, false);
        v16bf b2 = load_tile16(WTA, BU_, cn0, kbase + k0);
        acc[2] = __builtin_amdgcn_wmma_f32_16x16x32_bf16(false, a, false, b2, (short)0,
                                                         acc[2], false, false);
        v16bf b3 = load_tile16(WTB, BU_, cn0, kbase + k0);
        acc[3] = __builtin_amdgcn_wmma_f32_16x16x32_bf16(false, a, false, b3, (short)0,
                                                         acc[3], false, false);
      }
      if (kh) {                                  // upper-K wave publishes partials via LDS
#pragma unroll
        for (int hd = 0; hd < 4; ++hd)
#pragma unroll
          for (int v = 0; v < 8; ++v)
            red[p][hd][v][lane] = acc[hd][v];
      }
      __syncthreads();
      if (!kh) {                                 // lower-K wave reduces + blends
        const int half = lane >> 4;
        const int nc   = cn0 + (lane & 15);
        const float b1v = bff1[nc], b2v = bff2[nc];
        const float bav = bta[nc],  bbv = btb[nc];
#pragma unroll
        for (int v = 0; v < 8; ++v) {
          const int m = cm0 + v + 8 * half;      // batch row
          const float r0 = acc[0][v] + red[p][0][v][lane] + b1v;
          const float r1 = acc[1][v] + red[p][1][v][lane] + b2v;
          const float r2 = acc[2][v] + red[p][2][v][lane] + bav;
          const float r3 = acc[3][v] + red[p][3][v][lane] + bbv;
          const float f1 = fast_tanh(r0);
          const float f2 = fast_tanh(r1);
          const float s  = fast_sigmoid(r2 + r3);
          const float h  = f1 + s * (f2 - f1);   // ff1*(1-s) + s*ff2
          out[((size_t)m * T_ + t) * H_ + nc] = h;
          Z0[m * KZ_ + I_ + nc] = (__bf16)h;     // h feeds next step's concat
        }
      }
      if (t + 1 < T_) {                          // stage x[:, t+1, :] into concat buffer
#pragma unroll
        for (int i = 0; i < (B_ * I_) / (NWG_ * THREADS_); ++i) {
          const int e = gid + i * (NWG_ * THREADS_);
          const int b = e >> 8, c = e & (I_ - 1);
          Z0[b * KZ_ + c] = (__bf16)x[((size_t)b * T_ + (t + 1)) * I_ + c];
        }
      }
    }
    grid_sync(bcnt, bgen);
  }
}

// ---------------- host entry ----------------
extern "C" void kernel_launch(void* const* d_in, const int* in_sizes, int n_in,
                              void* d_out, int out_size, void* d_ws, size_t ws_size,
                              hipStream_t stream) {
  const float* x    = (const float*)d_in[0];
  const float* wb1  = (const float*)d_in[1];
  const float* bb1  = (const float*)d_in[2];
  const float* wb2  = (const float*)d_in[3];
  const float* bb2  = (const float*)d_in[4];
  const float* wff1 = (const float*)d_in[5];
  const float* bff1 = (const float*)d_in[6];
  const float* wff2 = (const float*)d_in[7];
  const float* bff2 = (const float*)d_in[8];
  const float* wta  = (const float*)d_in[9];
  const float* bta  = (const float*)d_in[10];
  const float* wtb  = (const float*)d_in[11];
  const float* btb  = (const float*)d_in[12];
  unsigned char* ws = (unsigned char*)d_ws;
  (void)in_sizes; (void)n_in; (void)out_size; (void)ws_size;

  cfc_prep<<<64, 256, 0, stream>>>(x, wb1, wb2, wff1, wff2, wta, wtb, ws);
  cfc_persistent<<<NWG_, THREADS_, 0, stream>>>(x, bb1, bb2, bff1, bff2, bta, btb,
                                                (float*)d_out, ws);
}
